// CPDNetwork_35115652612210
// MI455X (gfx1250) — matrix operand order repaired and checked by reading, not compiled, not run on hardware
//
#include <hip/hip_runtime.h>
#include <hip/hip_bf16.h>
#include <math.h>

// ---------------- problem constants (from reference) ----------------
#define NGRP 8
#define LGRP 384
#define KNN 30
#define NNODES (NGRP * LGRP)     // 3072
#define NEDGES (NNODES * KNN)    // 92160
#define NRBF 16
#define NPE 16
#define NAA 20
#define EPSV 1e-8f

typedef __bf16 bf16_t;
typedef __attribute__((ext_vector_type(4)))  __bf16 v4bf;
typedef __attribute__((ext_vector_type(8)))  __bf16 v8bf;
typedef __attribute__((ext_vector_type(16))) __bf16 v16bf;
typedef __attribute__((ext_vector_type(8)))  float  v8f;

// =====================================================================
// WMMA GEMM:  C[M,N] = act( A[M,K] * W[N,K]^T + bias )
// A, W in f32 (converted to bf16 into LDS), C in f32.
// Block: 256 threads = 8 waves. Block tile: 128(M) x (16*NT)(N), K-step 32.
// Requirements: M % 128 == 0 (all call sites satisfy this).
// LDS rows padded to 40 bf16 (80 B): 16B-aligned rows (b128 loads) and a
// bank stride of 20 words -> conflict-free column-of-16 fragment reads.
// =====================================================================
#define LDSPAD 40

template <int NT>
__global__ __launch_bounds__(256) void gemm_bf16_wmma(
    const float* __restrict__ A,
    const float* __restrict__ W,
    const float* __restrict__ bias,
    float* __restrict__ C,
    int M, int Nd, int Kd, int relu)
{
  constexpr int BN = 16 * NT;
  __shared__ bf16_t As[128][LDSPAD];   // 10 KB
  __shared__ bf16_t Bs[BN][LDSPAD];    // up to 5 KB

  const int tid  = threadIdx.x;
  const int wave = tid >> 5;
  const int lane = tid & 31;
  const int m0 = blockIdx.y * 128;
  const int n0 = blockIdx.x * BN;
  const bool k4 = (Kd & 3) == 0;
  (void)M;  // M % 128 == 0 guaranteed by caller

  v8f acc[NT];
#pragma unroll
  for (int t = 0; t < NT; ++t)
    acc[t] = (v8f){0.f, 0.f, 0.f, 0.f, 0.f, 0.f, 0.f, 0.f};

  // A staging map: 2 threads per row, 16 consecutive K each
  const int ar  = tid >> 1;
  const int acb = (tid & 1) * 16;
  const float* Arow = A + (size_t)(m0 + ar) * Kd;

  for (int k0 = 0; k0 < Kd; k0 += 32) {
    // CDNA5 prefetch of the next A K-slab (global_prefetch_b8)
    if (k0 + 32 < Kd) __builtin_prefetch(Arow + k0 + 32 + acb, 0, 1);

    // ---- stage A tile (128 x 32) ----
    if (k4) {
#pragma unroll
      for (int q = 0; q < 4; ++q) {
        int k = k0 + acb + q * 4;
        int kc = (k < Kd) ? k : (Kd >= 4 ? Kd - 4 : 0);
        float4 v = *(const float4*)(Arow + kc);     // always in-bounds
        bool ok = (k < Kd);
        v4bf p;
        p[0] = (bf16_t)(ok ? v.x : 0.f);
        p[1] = (bf16_t)(ok ? v.y : 0.f);
        p[2] = (bf16_t)(ok ? v.z : 0.f);
        p[3] = (bf16_t)(ok ? v.w : 0.f);
        *(v4bf*)&As[ar][acb + q * 4] = p;
      }
    } else {
#pragma unroll
      for (int c = 0; c < 16; ++c) {
        int k = k0 + acb + c;
        int kc = (k < Kd) ? k : (Kd - 1);
        float v = Arow[kc];                          // clamped, branchless
        As[ar][acb + c] = (bf16_t)((k < Kd) ? v : 0.f);
      }
    }

    // ---- stage B tile (BN x 32), n-major ----
    for (int t = tid; t < BN * 8; t += 256) {
      int n = t >> 3, q = t & 7;
      int gn = n0 + n;
      int k = k0 + (q & 3) * 4 + (q >> 2) * 16;      // 2 chunks of 16
      const float* Wr = W + (size_t)(gn < Nd ? gn : 0) * Kd;
      int kk = (q >> 2) * 16 + (q & 3) * 4;
      if (k4) {
        int kc = (k < Kd) ? k : (Kd >= 4 ? Kd - 4 : 0);
        float4 v = *(const float4*)(Wr + kc);
        bool ok = (k < Kd) && (gn < Nd);
        v4bf p;
        p[0] = (bf16_t)(ok ? v.x : 0.f);
        p[1] = (bf16_t)(ok ? v.y : 0.f);
        p[2] = (bf16_t)(ok ? v.z : 0.f);
        p[3] = (bf16_t)(ok ? v.w : 0.f);
        *(v4bf*)&Bs[n][kk] = p;
      } else {
#pragma unroll
        for (int c = 0; c < 4; ++c) {
          int kq = k + c;
          int kc = (kq < Kd) ? kq : (Kd - 1);
          float v = Wr[kc];
          Bs[n][kk + c] = (bf16_t)((kq < Kd && gn < Nd) ? v : 0.f);
        }
      }
    }
    __syncthreads();

    // ---- fragments: 2 x ds_load_b128 each ----
    // A (16x32 bf16): lane holds row wave*16 + lane%16; lanes 0-15 carry
    // K 0..7 / 16..23, lanes 16-31 carry K 8..15 / 24..31.
    const int row = (wave << 4) + (lane & 15);
    const int hf  = lane >> 4;
    v8bf a0 = *(const v8bf*)&As[row][hf * 8];
    v8bf a1 = *(const v8bf*)&As[row][16 + hf * 8];
    v16bf a = __builtin_shufflevector(a0, a1, 0, 1, 2, 3, 4, 5, 6, 7,
                                      8, 9, 10, 11, 12, 13, 14, 15);
    const int col = lane & 15;
#pragma unroll
    for (int t = 0; t < NT; ++t) {
      // B (32x16 bf16): lane holds column col; lanes 0-15 K=0..7 / 16..23,
      // lanes 16-31 K=8..15 / 24..31.
      v8bf b0 = *(const v8bf*)&Bs[t * 16 + col][hf * 8];
      v8bf b1 = *(const v8bf*)&Bs[t * 16 + col][16 + hf * 8];
      v16bf b = __builtin_shufflevector(b0, b1, 0, 1, 2, 3, 4, 5, 6, 7,
                                        8, 9, 10, 11, 12, 13, 14, 15);
      acc[t] = __builtin_amdgcn_wmma_f32_16x16x32_bf16(
          false, a, false, b, (short)0, acc[t], false, false);
    }
    __syncthreads();
  }

  // ---- epilogue: VGPR r holds M=r (lanes 0-15) / M=8+r (lanes 16-31) ----
  const int mrow = m0 + (wave << 4) + ((lane >> 4) << 3);
#pragma unroll
  for (int t = 0; t < NT; ++t) {
    int gn = n0 + t * 16 + (lane & 15);
    if (gn < Nd) {
      float bv = bias ? bias[gn] : 0.f;
#pragma unroll
      for (int r = 0; r < 8; ++r) {
        float v = acc[t][r] + bv;
        if (relu) v = fmaxf(v, 0.f);
        C[(size_t)(mrow + r) * Nd + gn] = v;
      }
    }
  }
}

// =====================================================================
// kNN within each group (LDS-cached positions, one block per group)
// =====================================================================
__global__ __launch_bounds__(LGRP) void knn_kernel(
    const float* __restrict__ pos, int* __restrict__ jidx)
{
  __shared__ float sp[LGRP][3];
  const int g = blockIdx.x;
  const int l = threadIdx.x;
  const float* p = pos + (long long)g * LGRP * 3;
  sp[l][0] = p[l * 3 + 0];
  sp[l][1] = p[l * 3 + 1];
  sp[l][2] = p[l * 3 + 2];
  __syncthreads();

  const float x = sp[l][0], y = sp[l][1], z = sp[l][2];
  unsigned mask[(LGRP + 31) / 32];
#pragma unroll
  for (int w = 0; w < (LGRP + 31) / 32; ++w) mask[w] = 0u;
  mask[l >> 5] |= 1u << (l & 31);  // exclude self

  for (int kk = 0; kk < KNN; ++kk) {
    float best = 3.4e38f;
    int bi = l;
    for (int m = 0; m < LGRP; ++m) {
      if (mask[m >> 5] & (1u << (m & 31))) continue;
      float dx = x - sp[m][0], dy = y - sp[m][1], dz = z - sp[m][2];
      float d2 = dx * dx + dy * dy + dz * dz;
      if (d2 < best) { best = d2; bi = m; }
    }
    mask[bi >> 5] |= 1u << (bi & 31);
    jidx[((long long)g * LGRP + l) * KNN + kk] = g * LGRP + bi;
  }
}

// =====================================================================
// Edge geometry: RBF(16)+pos-emb(16) -> edge_s[E,32]; unit evec vecpack
// =====================================================================
__global__ void edge_geom_kernel(const float* __restrict__ pos,
                                 const int* __restrict__ jidx,
                                 float* __restrict__ edge_s,
                                 float* __restrict__ evp)
{
  int e = blockIdx.x * blockDim.x + threadIdx.x;
  if (e >= NEDGES) return;
  int i = e / KNN;
  int j = jidx[e];
  float ex = pos[i * 3 + 0] - pos[j * 3 + 0];
  float ey = pos[i * 3 + 1] - pos[j * 3 + 1];
  float ez = pos[i * 3 + 2] - pos[j * 3 + 2];
  float D = sqrtf(ex * ex + ey * ey + ez * ez);
  const float sigma = 20.f / NRBF;
#pragma unroll
  for (int m = 0; m < NRBF; ++m) {
    float mu = 20.f * (float)m / (float)(NRBF - 1);
    float t = (D - mu) / sigma;
    edge_s[(long long)e * 32 + m] = expf(-t * t);
  }
  float d = (float)(i - j);
#pragma unroll
  for (int f = 0; f < NPE / 2; ++f) {
    float freq = expf(-(logf(10000.f) / NPE) * (float)(2 * f));
    float ang = d * freq;
    edge_s[(long long)e * 32 + 16 + f] = cosf(ang);
    edge_s[(long long)e * 32 + 24 + f] = sinf(ang);
  }
  float inv = (D > 0.f) ? 1.f / D : 0.f;
  evp[(long long)e * 3 + 0] = ex * inv;
  evp[(long long)e * 3 + 1] = ey * inv;
  evp[(long long)e * 3 + 2] = ez * inv;
}

// =====================================================================
// Local frame basis R (3x3, columns e1,e2,e3), row-major
// =====================================================================
__global__ void basis_kernel(const float* __restrict__ ca,
                             const float* __restrict__ cc,
                             const float* __restrict__ nn,
                             float* __restrict__ R)
{
  int n = blockIdx.x * blockDim.x + threadIdx.x;
  if (n >= NNODES) return;
  float u1[3], e1[3], u2[3], e2[3], e3[3];
  for (int a = 0; a < 3; ++a) u1[a] = cc[n * 3 + a] - ca[n * 3 + a];
  float n1 = sqrtf(u1[0] * u1[0] + u1[1] * u1[1] + u1[2] * u1[2]);
  float i1 = (n1 > 0.f) ? 1.f / n1 : 0.f;
  for (int a = 0; a < 3; ++a) e1[a] = u1[a] * i1;
  for (int a = 0; a < 3; ++a) u2[a] = nn[n * 3 + a] - ca[n * 3 + a];
  float dp = u2[0] * e1[0] + u2[1] * e1[1] + u2[2] * e1[2];
  for (int a = 0; a < 3; ++a) u2[a] -= dp * e1[a];
  float n2 = sqrtf(u2[0] * u2[0] + u2[1] * u2[1] + u2[2] * u2[2]);
  float i2 = (n2 > 0.f) ? 1.f / n2 : 0.f;
  for (int a = 0; a < 3; ++a) e2[a] = u2[a] * i2;
  e3[0] = e1[1] * e2[2] - e1[2] * e2[1];
  e3[1] = e1[2] * e2[0] - e1[0] * e2[2];
  e3[2] = e1[0] * e2[1] - e1[1] * e2[0];
  for (int a = 0; a < 3; ++a) {
    float v1 = isfinite(e1[a]) ? e1[a] : 0.f;
    float v2 = isfinite(e2[a]) ? e2[a] : 0.f;
    float v3 = isfinite(e3[a]) ? e3[a] : 0.f;
    R[(long long)n * 9 + a * 3 + 0] = v1;
    R[(long long)n * 9 + a * 3 + 1] = v2;
    R[(long long)n * 9 + a * 3 + 2] = v3;
  }
}

// repack (M,C,3) -> vecpack (3M,C)
__global__ void repack_v_kernel(const float* __restrict__ in,
                                float* __restrict__ out, int M, int C)
{
  long long idx = (long long)blockIdx.x * blockDim.x + threadIdx.x;
  long long tot = (long long)M * C * 3;
  if (idx >= tot) return;
  int i = (int)(idx % 3);
  long long r = idx / 3;
  int c = (int)(r % C);
  int m = (int)(r / C);
  out[((long long)m * 3 + i) * C + c] = in[(long long)m * C * 3 + c * 3 + i];
}

// rotate vecpack: trans=1 -> rot_in (R^T), trans=0 -> rot_out (R)
__global__ void rot_kernel(const float* __restrict__ Rb, int rdiv,
                           const float* __restrict__ in,
                           float* __restrict__ out, int M, int C, int trans)
{
  long long idx = (long long)blockIdx.x * blockDim.x + threadIdx.x;
  long long tot = (long long)M * C;
  if (idx >= tot) return;
  int c = (int)(idx % C);
  int m = (int)(idx / C);
  const float* Rm = Rb + (long long)(m / rdiv) * 9;
  float v0 = in[((long long)m * 3 + 0) * C + c];
  float v1 = in[((long long)m * 3 + 1) * C + c];
  float v2 = in[((long long)m * 3 + 2) * C + c];
#pragma unroll
  for (int i = 0; i < 3; ++i) {
    float r0 = trans ? Rm[0 * 3 + i] : Rm[i * 3 + 0];
    float r1 = trans ? Rm[1 * 3 + i] : Rm[i * 3 + 1];
    float r2 = trans ? Rm[2 * 3 + i] : Rm[i * 3 + 2];
    out[((long long)m * 3 + i) * C + c] = r0 * v0 + r1 * v1 + r2 * v2;
  }
}

// cat[M, si+h] = [ s | ||vh|| ]
__global__ void vn_cat_kernel(const float* __restrict__ s,
                              const float* __restrict__ vh,
                              float* __restrict__ cat, int M, int si, int h)
{
  long long idx = (long long)blockIdx.x * blockDim.x + threadIdx.x;
  int cols = si + h;
  long long tot = (long long)M * cols;
  if (idx >= tot) return;
  int c = (int)(idx % cols);
  int m = (int)(idx / cols);
  if (c < si) {
    cat[idx] = s[(long long)m * si + c];
  } else {
    int hh = c - si;
    float a = vh[((long long)m * 3 + 0) * h + hh];
    float b = vh[((long long)m * 3 + 1) * h + hh];
    float d = vh[((long long)m * 3 + 2) * h + hh];
    cat[idx] = sqrtf(a * a + b * b + d * d + EPSV);
  }
}

// v *= sigmoid(||v||) per channel
__global__ void vgate_kernel(float* __restrict__ vp, int M, int C)
{
  long long idx = (long long)blockIdx.x * blockDim.x + threadIdx.x;
  long long tot = (long long)M * C;
  if (idx >= tot) return;
  int c = (int)(idx % C);
  int m = (int)(idx / C);
  float a = vp[((long long)m * 3 + 0) * C + c];
  float b = vp[((long long)m * 3 + 1) * C + c];
  float d = vp[((long long)m * 3 + 2) * C + c];
  float g = 1.f / (1.f + expf(-sqrtf(a * a + b * b + d * d + EPSV)));
  vp[((long long)m * 3 + 0) * C + c] = a * g;
  vp[((long long)m * 3 + 1) * C + c] = b * g;
  vp[((long long)m * 3 + 2) * C + c] = d * g;
}

__device__ inline float block_reduce_sum(float v, float* sh)
{
  int t = threadIdx.x;
  sh[t] = v;
  __syncthreads();
  for (int s = blockDim.x >> 1; s > 0; s >>= 1) {
    if (t < s) sh[t] += sh[t + s];
    __syncthreads();
  }
  float r = sh[0];
  __syncthreads();
  return r;
}

// scalar LayerNorm + vector RMS-over-channels norm, in place, block/row
__global__ __launch_bounds__(128) void svln_kernel(
    float* __restrict__ s, float* __restrict__ vp,
    const float* __restrict__ gamma, const float* __restrict__ beta,
    int S, int C)
{
  __shared__ float sh[128];
  const int m = blockIdx.x;
  float acc = 0.f;
  for (int c = threadIdx.x; c < S; c += blockDim.x) acc += s[(long long)m * S + c];
  float mu = block_reduce_sum(acc, sh) / (float)S;
  acc = 0.f;
  for (int c = threadIdx.x; c < S; c += blockDim.x) {
    float d = s[(long long)m * S + c] - mu;
    acc += d * d;
  }
  float var = block_reduce_sum(acc, sh) / (float)S;
  float inv = rsqrtf(var + 1e-5f);
  for (int c = threadIdx.x; c < S; c += blockDim.x) {
    s[(long long)m * S + c] =
        (s[(long long)m * S + c] - mu) * inv * gamma[c] + beta[c];
  }
  if (vp) {
    acc = 0.f;
    for (int t = threadIdx.x; t < 3 * C; t += blockDim.x) {
      int i = t / C, c = t % C;
      float v = vp[((long long)m * 3 + i) * C + c];
      acc += v * v;
    }
    float tot = block_reduce_sum(acc, sh);
    float scl = rsqrtf(tot / (float)C + EPSV);
    for (int t = threadIdx.x; t < 3 * C; t += blockDim.x) {
      int i = t / C, c = t % C;
      vp[((long long)m * 3 + i) * C + c] *= scl;
    }
  }
}

// gather scalar message inputs: [sj | es | (hseq) | si]
__global__ void gather_s_kernel(const float* __restrict__ s,
                                const float* __restrict__ encs,
                                const float* __restrict__ es,
                                const float* __restrict__ aa,
                                const int* __restrict__ seq,
                                const int* __restrict__ jidx,
                                float* __restrict__ out, int dec)
{
  const int cols = dec ? 340 : 320;
  long long idx = (long long)blockIdx.x * blockDim.x + threadIdx.x;
  long long tot = (long long)NEDGES * cols;
  if (idx >= tot) return;
  int c = (int)(idx % cols);
  int e = (int)(idx / cols);
  int i = e / KNN;
  int j = jidx[e];
  bool fwd = (j < i);
  float val;
  if (c < 128) {
    val = (!dec || fwd) ? s[(long long)j * 128 + c] : encs[(long long)j * 128 + c];
  } else if (c < 192) {
    val = es[(long long)e * 64 + (c - 128)];
  } else if (dec && c < 212) {
    val = fwd ? aa[(long long)seq[j] * NAA + (c - 192)] : 0.f;
  } else {
    int cc = c - (dec ? 212 : 192);
    val = (!dec || fwd) ? s[(long long)i * 128 + cc] : encs[(long long)i * 128 + cc];
  }
  out[idx] = val;
}

// gather vector message inputs: channels [vj(32) | ev(16) | vi(32)]
__global__ void gather_v_kernel(const float* __restrict__ vp,
                                const float* __restrict__ encvp,
                                const float* __restrict__ evp,
                                const int* __restrict__ jidx,
                                float* __restrict__ out, int dec)
{
  long long idx = (long long)blockIdx.x * blockDim.x + threadIdx.x;
  long long tot = (long long)NEDGES * 3 * 80;
  if (idx >= tot) return;
  int col = (int)(idx % 80);
  long long r = idx / 80;
  int d = (int)(r % 3);
  int e = (int)(r / 3);
  int i = e / KNN;
  int j = jidx[e];
  bool fwd = (j < i);
  float val;
  if (col < 32) {
    const float* src = (!dec || fwd) ? vp : encvp;
    val = src[((long long)j * 3 + d) * 32 + col];
  } else if (col < 48) {
    val = evp[((long long)e * 3 + d) * 16 + (col - 32)];
  } else {
    const float* src = (!dec || fwd) ? vp : encvp;
    val = src[((long long)i * 3 + d) * 32 + (col - 48)];
  }
  out[idx] = val;
}

// segment means: edges of node n are contiguous [n*K, (n+1)*K)
__global__ void agg_s_kernel(const float* __restrict__ ms, float* __restrict__ ds, int S)
{
  long long idx = (long long)blockIdx.x * blockDim.x + threadIdx.x;
  long long tot = (long long)NNODES * S;
  if (idx >= tot) return;
  int c = (int)(idx % S);
  int n = (int)(idx / S);
  float acc = 0.f;
  for (int kk = 0; kk < KNN; ++kk)
    acc += ms[((long long)n * KNN + kk) * S + c];
  ds[idx] = acc / (float)KNN;
}

__global__ void agg_v_kernel(const float* __restrict__ mv, float* __restrict__ dv, int C)
{
  long long idx = (long long)blockIdx.x * blockDim.x + threadIdx.x;
  long long tot = (long long)NNODES * 3 * C;
  if (idx >= tot) return;
  int c = (int)(idx % C);
  long long r = idx / C;
  int i = (int)(r % 3);
  int n = (int)(r / 3);
  float acc = 0.f;
  for (int kk = 0; kk < KNN; ++kk)
    acc += mv[(((long long)n * KNN + kk) * 3 + i) * C + c];
  dv[((long long)n * 3 + i) * C + c] = acc / (float)KNN;
}

__global__ void add_kernel(const float* __restrict__ a, const float* __restrict__ b,
                           float* __restrict__ c, long long n)
{
  long long idx = (long long)blockIdx.x * blockDim.x + threadIdx.x;
  if (idx < n) c[idx] = a[idx] + b[idx];
}

// =====================================================================
// Host orchestration
// =====================================================================
struct GvpW { const float *bs, *wh, *ws, *wv; };
struct LnW  { const float *b, *g; };
struct ConvW { GvpW ff1, ff2; LnW ln1, ln2; GvpW m1, m2, m3; };

static inline unsigned nb(long long n, int b) { return (unsigned)((n + b - 1) / b); }

static void gemm(hipStream_t st, const float* A, const float* W, const float* bias,
                 float* C, int M, int Nd, int Kd, int relu)
{
  if (Nd >= 64) {
    dim3 g(nb(Nd, 64), nb(M, 128));
    gemm_bf16_wmma<4><<<g, 256, 0, st>>>(A, W, bias, C, M, Nd, Kd, relu);
  } else if (Nd >= 32) {
    dim3 g(nb(Nd, 32), nb(M, 128));
    gemm_bf16_wmma<2><<<g, 256, 0, st>>>(A, W, bias, C, M, Nd, Kd, relu);
  } else {
    dim3 g(nb(Nd, 16), nb(M, 128));
    gemm_bf16_wmma<1><<<g, 256, 0, st>>>(A, W, bias, C, M, Nd, Kd, relu);
  }
}

static void run_gvp(hipStream_t st, int M, const float* s_in, const float* vp_in,
                    const GvpW& w, int si, int vi, int so, int vo,
                    bool sact, bool vact, float* s_out, float* vp_out,
                    float* vh_tmp, float* cat_tmp)
{
  int h = (vo > 0) ? (vi > vo ? vi : vo) : vi;
  gemm(st, vp_in, w.wh, nullptr, vh_tmp, 3 * M, h, vi, 0);
  vn_cat_kernel<<<nb((long long)M * (si + h), 256), 256, 0, st>>>(
      s_in, vh_tmp, cat_tmp, M, si, h);
  gemm(st, cat_tmp, w.ws, w.bs, s_out, M, so, si + h, sact ? 1 : 0);
  if (vo > 0) {
    gemm(st, vh_tmp, w.wv, nullptr, vp_out, 3 * M, vo, h, 0);
    if (vact)
      vgate_kernel<<<nb((long long)M * vo, 256), 256, 0, st>>>(vp_out, M, vo);
  }
}

extern "C" void kernel_launch(void* const* d_in, const int* in_sizes, int n_in,
                              void* d_out, int out_size, void* d_ws, size_t ws_size,
                              hipStream_t stream)
{
  (void)in_sizes; (void)n_in; (void)out_size; (void)ws_size;
  // ---- parameter leaves in JAX pytree order (dict keys sorted) ----
  int ip = 0;
  auto nxt = [&]() -> const float* { return (const float*)d_in[ip++]; };
  auto rdGvp = [&](bool hasWv) {
    GvpW g; g.bs = nxt(); g.wh = nxt(); g.ws = nxt(); g.wv = hasWv ? nxt() : nullptr;
    return g;
  };
  auto rdLn = [&]() { LnW l; l.b = nxt(); l.g = nxt(); return l; };
  auto rdConv = [&]() {
    ConvW c;
    c.ff1 = rdGvp(true); c.ff2 = rdGvp(true);
    c.ln1 = rdLn(); c.ln2 = rdLn();
    c.m1 = rdGvp(true); c.m2 = rdGvp(true); c.m3 = rdGvp(true);
    return c;
  };
  const float* aa_embed = nxt();                 // 'aa_embed'
  const float* d1b = nxt(); const float* d1w = nxt();
  const float* d2b = nxt(); const float* d2w = nxt();
  ConvW decL[3]; for (int l = 0; l < 3; ++l) decL[l] = rdConv();   // 'dec'
  ConvW encL[3]; for (int l = 0; l < 3; ++l) encL[l] = rdConv();   // 'enc'
  GvpW we_g1 = rdGvp(true), we_g2 = rdGvp(true); LnW we_ln = rdLn(); // 'wedge'
  GvpW wn_g1 = rdGvp(true), wn_g2 = rdGvp(true); LnW wn_ln = rdLn(); // 'wnode'
  GvpW wo_g = rdGvp(false); LnW wo_ln = rdLn();                      // 'wout'
  const float* node_s = nxt();
  const float* node_v = nxt();
  const float* pos_CA = nxt();
  const float* pos_C  = nxt();
  const float* pos_N  = nxt();
  const int*   seq    = (const int*)d_in[ip++];

  // ---- scratch bump allocator ----
  char* wp = (char*)d_ws;
  auto allocf = [&](long long elems) -> float* {
    float* p = (float*)wp;
    wp += ((elems * 4 + 255) & ~255LL);
    return p;
  };
  int* jidx = (int*)allocf(NEDGES);
  float* Rb       = allocf((long long)NNODES * 9);
  float* edge_s32 = allocf((long long)NEDGES * 32);
  float* evp0     = allocf((long long)NEDGES * 3);
  float* evprot   = allocf((long long)NEDGES * 3);
  float* esA   = allocf((long long)NEDGES * 64);
  float* esB   = allocf((long long)NEDGES * 64);
  float* edvpA = allocf((long long)NEDGES * 3 * 16);
  float* edvpB = allocf((long long)NEDGES * 3 * 16);
  float* sA    = allocf((long long)NNODES * 128);
  float* sB    = allocf((long long)NNODES * 128);
  float* vpA   = allocf((long long)NNODES * 3 * 32);
  float* vpB   = allocf((long long)NNODES * 3 * 32);
  float* enc_s  = allocf((long long)NNODES * 128);
  float* enc_vp = allocf((long long)NNODES * 3 * 32);
  float* nvpack = allocf((long long)NNODES * 3 * 3);
  float* nvrot  = allocf((long long)NNODES * 3 * 3);
  float* nodevh  = allocf((long long)NNODES * 3 * 64);
  float* nodecat = allocf((long long)NNODES * 576);
  float* s_res  = allocf((long long)NNODES * 128);
  float* vp_res = allocf((long long)NNODES * 3 * 32);
  float* vrot   = allocf((long long)NNODES * 3 * 64);
  float* fs   = allocf((long long)NNODES * 512);
  float* fvp  = allocf((long long)NNODES * 3 * 64);
  float* fs2  = allocf((long long)NNODES * 128);
  float* fvp2 = allocf((long long)NNODES * 3 * 32);
  float* dsb  = allocf((long long)NNODES * 128);
  float* dvp  = allocf((long long)NNODES * 3 * 32);
  float* msin = allocf((long long)NEDGES * 340);
  float* mvin = allocf((long long)NEDGES * 3 * 80);
  float* evh  = allocf((long long)NEDGES * 3 * 80);
  float* ecat = allocf((long long)NEDGES * 420);
  float* ms1  = allocf((long long)NEDGES * 128);
  float* mv1  = allocf((long long)NEDGES * 3 * 32);
  float* ms2  = allocf((long long)NEDGES * 128);
  float* mv2  = allocf((long long)NEDGES * 3 * 32);
  float* outS = allocf((long long)NNODES * 128);
  float* h1b  = allocf((long long)NNODES * 128);

  hipStream_t st = stream;

  // ---- graph geometry ----
  knn_kernel<<<NGRP, LGRP, 0, st>>>(pos_CA, jidx);
  basis_kernel<<<nb(NNODES, 256), 256, 0, st>>>(pos_CA, pos_C, pos_N, Rb);
  edge_geom_kernel<<<nb(NEDGES, 256), 256, 0, st>>>(pos_CA, jidx, edge_s32, evp0);

  // ---- wnode sv_block ----
  repack_v_kernel<<<nb((long long)NNODES * 9, 256), 256, 0, st>>>(node_v, nvpack, NNODES, 3);
  rot_kernel<<<nb((long long)NNODES * 3, 256), 256, 0, st>>>(Rb, 1, nvpack, nvrot, NNODES, 3, 1);
  run_gvp(st, NNODES, node_s, nvrot, wn_g1, 6, 3, 128, 32, false, false, sA, vpA, nodevh, nodecat);
  svln_kernel<<<NNODES, 128, 0, st>>>(sA, vpA, wn_ln.g, wn_ln.b, 128, 32);
  run_gvp(st, NNODES, sA, vpA, wn_g2, 128, 32, 128, 32, false, false, sB, vpB, nodevh, nodecat);
  rot_kernel<<<nb((long long)NNODES * 32, 256), 256, 0, st>>>(Rb, 1, vpB, vpA, NNODES, 32, 0);
  float* cur_s = sB;
  float* cur_vp = vpA;

  // ---- wedge sv_block ----
  rot_kernel<<<nb((long long)NEDGES * 1, 256), 256, 0, st>>>(Rb, KNN, evp0, evprot, NEDGES, 1, 1);
  run_gvp(st, NEDGES, edge_s32, evprot, we_g1, 32, 1, 64, 16, false, false, esA, edvpA, evh, ecat);
  svln_kernel<<<NEDGES, 128, 0, st>>>(esA, edvpA, we_ln.g, we_ln.b, 64, 16);
  run_gvp(st, NEDGES, esA, edvpA, we_g2, 64, 16, 64, 16, false, false, esB, edvpB, evh, ecat);
  rot_kernel<<<nb((long long)NEDGES * 16, 256), 256, 0, st>>>(Rb, KNN, edvpB, edvpA, NEDGES, 16, 0);
  float* es_buf = esB;
  float* ev_buf = edvpA;

  // ---- one message-passing + feed-forward layer ----
  auto do_layer = [&](const ConvW& p, int dec_mode) {
    int si1 = dec_mode ? 340 : 320;
    gather_s_kernel<<<nb((long long)NEDGES * si1, 256), 256, 0, st>>>(
        cur_s, enc_s, es_buf, aa_embed, seq, jidx, msin, dec_mode);
    gather_v_kernel<<<nb((long long)NEDGES * 3 * 80, 256), 256, 0, st>>>(
        cur_vp, enc_vp, ev_buf, jidx, mvin, dec_mode);
    run_gvp(st, NEDGES, msin, mvin, p.m1, si1, 80, 128, 32, true, true, ms1, mv1, evh, ecat);
    run_gvp(st, NEDGES, ms1, mv1, p.m2, 128, 32, 128, 32, true, true, ms2, mv2, evh, ecat);
    run_gvp(st, NEDGES, ms2, mv2, p.m3, 128, 32, 128, 32, false, false, ms1, mv1, evh, ecat);
    agg_s_kernel<<<nb((long long)NNODES * 128, 256), 256, 0, st>>>(ms1, dsb, 128);
    agg_v_kernel<<<nb((long long)NNODES * 3 * 32, 256), 256, 0, st>>>(mv1, dvp, 32);
    // ff_res
    add_kernel<<<nb((long long)NNODES * 128, 256), 256, 0, st>>>(cur_s, dsb, s_res, (long long)NNODES * 128);
    add_kernel<<<nb((long long)NNODES * 96, 256), 256, 0, st>>>(cur_vp, dvp, vp_res, (long long)NNODES * 3 * 32);
    svln_kernel<<<NNODES, 128, 0, st>>>(s_res, vp_res, p.ln1.g, p.ln1.b, 128, 32);
    rot_kernel<<<nb((long long)NNODES * 32, 256), 256, 0, st>>>(Rb, 1, vp_res, vrot, NNODES, 32, 1);
    run_gvp(st, NNODES, s_res, vrot, p.ff1, 128, 32, 512, 64, true, true, fs, fvp, nodevh, nodecat);
    run_gvp(st, NNODES, fs, fvp, p.ff2, 512, 64, 128, 32, false, false, fs2, fvp2, nodevh, nodecat);
    rot_kernel<<<nb((long long)NNODES * 32, 256), 256, 0, st>>>(Rb, 1, fvp2, vrot, NNODES, 32, 0);
    float* ns  = (cur_s == sA) ? sB : sA;
    float* nvp = (cur_vp == vpA) ? vpB : vpA;
    add_kernel<<<nb((long long)NNODES * 128, 256), 256, 0, st>>>(s_res, fs2, ns, (long long)NNODES * 128);
    add_kernel<<<nb((long long)NNODES * 96, 256), 256, 0, st>>>(vp_res, vrot, nvp, (long long)NNODES * 3 * 32);
    svln_kernel<<<NNODES, 128, 0, st>>>(ns, nvp, p.ln2.g, p.ln2.b, 128, 32);
    cur_s = ns;
    cur_vp = nvp;
  };

  for (int l = 0; l < 3; ++l) do_layer(encL[l], 0);
  hipMemcpyAsync(enc_s, cur_s, (long long)NNODES * 128 * 4, hipMemcpyDeviceToDevice, st);
  hipMemcpyAsync(enc_vp, cur_vp, (long long)NNODES * 3 * 32 * 4, hipMemcpyDeviceToDevice, st);
  for (int l = 0; l < 3; ++l) do_layer(decL[l], 1);

  // ---- output head ----
  svln_kernel<<<NNODES, 128, 0, st>>>(cur_s, cur_vp, wo_ln.g, wo_ln.b, 128, 32);
  rot_kernel<<<nb((long long)NNODES * 32, 256), 256, 0, st>>>(Rb, 1, cur_vp, vrot, NNODES, 32, 1);
  run_gvp(st, NNODES, cur_s, vrot, wo_g, 128, 32, 128, 0, true, false, outS, nullptr, nodevh, nodecat);
  gemm(st, outS, d1w, d1b, h1b, NNODES, 128, 128, 1);
  gemm(st, h1b, d2w, d2b, (float*)d_out, NNODES, 20, 128, 0);
}
